// MyModule_36661840839309
// MI455X (gfx1250) — compile-verified, matrix operand order checked by reading, not run on hardware
//
#include <hip/hip_runtime.h>
#include <math.h>

// ---------------------------------------------------------------------------
// Hierarchical BiGRU (word BiGRU -> pool -> sentence BiGRU -> pool -> heads)
// for MI455X / gfx1250, bf16 WMMA (v_wmma_f32_16x16x32_bf16) GEMMs.
//
// Per GRU step (per direction) one dense GEMM:
//   A = [x_t | h_{t-1}]  (M x Kpad, bf16),  B = [[W_ih^T,0],[0,W_hh^T]] (Kpad x 1216, bf16)
//   C = GH (M x 1216, f32):  cols 0..599 = gx (+b_ih), cols 608..1207 = gh (+b_hh)
// then an elementwise gate kernel updates h, writes bf16 h back into A, and
// accumulates the masked mean-pool on the fly.
// ---------------------------------------------------------------------------

#define NSEQ   2560      // N_DOCS * DOC_TRUNC
#define WORD_T 50
#define EMB_D  300
#define HID    200
#define NDOC   256
#define NSENT  10
#define GN     1216      // padded gate width: 600 gx | 8 pad | 600 gh
#define LDA_W  544       // word A: 320 (emb pad) + 224 (h pad)
#define LDA_S  640       // sent A: 416 (sent-vec pad) + 224 (h pad)
#define XS_LD  416       // padded sentence-vector width (2H=400 -> 416)
#define LPAD   40        // LDS row stride (ushorts): 80B -> 20-bank stride

typedef __bf16 v16bf __attribute__((ext_vector_type(16)));
typedef float  v8f   __attribute__((ext_vector_type(8)));

static __device__ __forceinline__ unsigned short f2bf(float f) {
    unsigned u = __builtin_bit_cast(unsigned, f);
    u += 0x7FFFu + ((u >> 16) & 1u);           // round-to-nearest-even
    return (unsigned short)(u >> 16);
}
static __device__ __forceinline__ __bf16 bfbits(unsigned short s) {
    return __builtin_bit_cast(__bf16, s);
}
static __device__ __forceinline__ float sigmoidf_(float x) {
    return 1.0f / (1.0f + expf(-x));
}

// ---------------------------------------------------------------------------
// WMMA GEMM:  C[M x GN] = A[M x (32*kTiles)] * B[(32*kTiles) x GN] + bias[GN]
// Block = 256 threads = 8 waves. Tile: BM=128 (16 rows/wave), BN=64 (4 wmma
// tiles per wave). A,B are bf16 (raw u16); C is f32.
// M must be a multiple of 128 (true for all launches here: 2560, 256).
//
// LDS layouts (row stride LPAD=40 ushorts to avoid 8-way bank conflicts):
//   Ash[128][k 0..31]          : per-lane A fragment = two contiguous 16B runs
//   Bsh[n 0..63][p=perm(k)]    : per-lane B fragment = one contiguous 32B run
// perm(k) packs each lane-half's K values contiguously:
//   lanes 0-15 need k {0..7,16..23} -> p 0..15 ; lanes 16-31 k {8..15,24..31} -> p 16..31
// ---------------------------------------------------------------------------
__global__ __launch_bounds__(256) void gemm_bf16_wmma(
    const unsigned short* __restrict__ A, int lda, int M,
    const unsigned short* __restrict__ B,
    const float* __restrict__ bias,
    float* __restrict__ C, int kTiles)
{
    (void)M;
    __shared__ __align__(16) unsigned short Ash[128 * LPAD];
    __shared__ __align__(16) unsigned short Bsh[64 * LPAD];

    const int tid    = threadIdx.x;
    const int wave   = tid >> 5;
    const int lane   = tid & 31;
    const int lane15 = lane & 15;
    const int kb     = (lane < 16) ? 0 : 8;   // A-fragment k base
    const int boff   = (lane < 16) ? 0 : 16;  // B-fragment p base
    const int bm     = blockIdx.x * 128;
    const int bn     = blockIdx.y * 64;

    // B staging: each thread owns one k row (bk) x 8-wide n chunk (bn8)
    const int bk  = tid >> 3;                 // 0..31
    const int bn8 = (tid & 7) * 8;            // 0..56
    const int bp  = ((bk & 8) ? 16 : 0) | ((bk & 16) ? 8 : 0) | (bk & 7);

    v8f acc[4] = {};

    for (int kt = 0; kt < kTiles; ++kt) {
        const int k0 = kt * 32;

        // ---- stage A tile: 128x32 bf16, two 16B vector loads per thread ----
        #pragma unroll
        for (int i = 0; i < 2; ++i) {
            int idx = tid * 2 + i;
            int r = idx >> 2, q = idx & 3;    // row 0..127, 16B chunk 0..3
            const uint4 v = *(const uint4*)(A + (size_t)(bm + r) * lda + k0 + q * 8);
            *(uint4*)(&Ash[r * LPAD + q * 8]) = v;
        }
        // ---- stage B tile: 32x64 bf16, one 16B load, transpose+permute ----
        {
            const uint4 v = *(const uint4*)(B + (size_t)(k0 + bk) * GN + bn + bn8);
            const unsigned int w[4] = {v.x, v.y, v.z, v.w};
            #pragma unroll
            for (int m = 0; m < 4; ++m) {
                Bsh[(bn8 + 2 * m)     * LPAD + bp] = (unsigned short)(w[m] & 0xffffu);
                Bsh[(bn8 + 2 * m + 1) * LPAD + bp] = (unsigned short)(w[m] >> 16);
            }
        }
        // prefetch next k-tile while this one is consumed
        if (kt + 1 < kTiles) {
            __builtin_prefetch(A + (size_t)(bm + (tid >> 1)) * lda + k0 + 32, 0, 0);
            __builtin_prefetch(B + (size_t)(k0 + 32 + bk) * GN + bn + bn8, 0, 0);
        }
        __syncthreads();

        // ---- A fragment: two contiguous 16B runs per lane ----
        v16bf af;
        {
            const int abase = (16 * wave + lane15) * LPAD + kb;
            #pragma unroll
            for (int e = 0; e < 8; ++e) af[e]     = bfbits(Ash[abase + e]);
            #pragma unroll
            for (int e = 0; e < 8; ++e) af[8 + e] = bfbits(Ash[abase + 16 + e]);
        }
        // ---- 4 B fragments / 4 WMMAs across the 64-wide N strip ----
        #pragma unroll
        for (int c4 = 0; c4 < 4; ++c4) {
            v16bf bfv;
            const int bbase = (16 * c4 + lane15) * LPAD + boff;
            #pragma unroll
            for (int e = 0; e < 16; ++e) bfv[e] = bfbits(Bsh[bbase + e]);
            acc[c4] = __builtin_amdgcn_wmma_f32_16x16x32_bf16(
                false, af, false, bfv, (short)0, acc[c4], false, false);
        }
        __syncthreads();
    }

    // --- epilogue: C/D layout -> row = v + (lane<16?0:8), col = lane%16 ---
    const int rofs = (lane < 16) ? 0 : 8;
    #pragma unroll
    for (int c4 = 0; c4 < 4; ++c4) {
        int n    = bn + 16 * c4 + lane15;
        float bv = bias[n];
        #pragma unroll
        for (int v = 0; v < 8; ++v) {
            int r = bm + 16 * wave + rofs + v;
            C[r * GN + n] = acc[c4][v] + bv;
        }
    }
}

// ---------------------------------------------------------------------------
// Elementwise / setup kernels
// ---------------------------------------------------------------------------
__global__ void zero_f32(float* p, int n) {
    int i = blockIdx.x * blockDim.x + threadIdx.x;
    if (i < n) p[i] = 0.0f;
}
__global__ void zero_u16(unsigned short* p, int n) {
    int i = blockIdx.x * blockDim.x + threadIdx.x;
    if (i < n) p[i] = 0;
}
__global__ void compute_lens(const int* __restrict__ x, int* __restrict__ lens) {
    int i = blockIdx.x * blockDim.x + threadIdx.x;
    if (i >= NSEQ) return;
    int c = 0;
    for (int t = 0; t < WORD_T; ++t) c += (x[i * WORD_T + t] > 0) ? 1 : 0;
    lens[i] = c;
}
// word-level combined weight: 544 x 1216 bf16
__global__ void fill_wc_word(const float* __restrict__ wih, const float* __restrict__ whh,
                             unsigned short* __restrict__ Wc) {
    int idx = blockIdx.x * blockDim.x + threadIdx.x;
    if (idx >= LDA_W * GN) return;
    int r = idx / GN, c = idx % GN;
    float v = 0.0f;
    if (r < EMB_D && c < 600)                       v = wih[c * EMB_D + r];
    else if (r >= 320 && r < 520 && c >= 608 && c < 1208)
                                                    v = whh[(c - 608) * HID + (r - 320)];
    Wc[idx] = f2bf(v);
}
// sentence-level combined weight: 640 x 1216 bf16
__global__ void fill_wc_sent(const float* __restrict__ sih, const float* __restrict__ shh,
                             unsigned short* __restrict__ Wc) {
    int idx = blockIdx.x * blockDim.x + threadIdx.x;
    if (idx >= LDA_S * GN) return;
    int r = idx / GN, c = idx % GN;
    float v = 0.0f;
    if (r < 400 && c < 600)                         v = sih[c * 400 + r];
    else if (r >= XS_LD && r < XS_LD + HID && c >= 608 && c < 1208)
                                                    v = shh[(c - 608) * HID + (r - XS_LD)];
    Wc[idx] = f2bf(v);
}
__global__ void fill_bias(const float* __restrict__ bi, const float* __restrict__ bh,
                          float* __restrict__ B) {
    int c = blockIdx.x * blockDim.x + threadIdx.x;
    if (c >= GN) return;
    float v = 0.0f;
    if (c < 600)                   v = bi[c];
    else if (c >= 608 && c < 1208) v = bh[c - 608];
    B[c] = v;
}
// gather emb[x[seq, t_eff]] (bf16) into word A-matrix cols 0..299
__global__ void gather_word(const int* __restrict__ x, const float* __restrict__ emb,
                            unsigned short* __restrict__ A, int t_eff) {
    int idx = blockIdx.x * blockDim.x + threadIdx.x;
    if (idx >= NSEQ * EMB_D) return;
    int seq = idx / EMB_D, k = idx % EMB_D;
    int tok = x[seq * WORD_T + t_eff];
    A[seq * LDA_W + k] = f2bf(emb[tok * EMB_D + k]);
}
// gather pooled sentence vector into sentence A-matrix cols 0..415
__global__ void gather_sent(const unsigned short* __restrict__ Xs,
                            unsigned short* __restrict__ A, int t_eff) {
    int idx = blockIdx.x * blockDim.x + threadIdx.x;
    if (idx >= NDOC * XS_LD) return;
    int doc = idx / XS_LD, k = idx % XS_LD;
    A[doc * LDA_S + k] = Xs[(doc * NSENT + t_eff) * XS_LD + k];
}
// GRU gates + h update + bf16 writeback into A + masked mean-pool accumulate
__global__ void gru_gate(const float* __restrict__ GH, float* __restrict__ H,
                         unsigned short* __restrict__ A, int lda, int hcol,
                         float* __restrict__ Sacc, const int* __restrict__ lens,
                         int t_eff, int dir, int M) {
    int idx = blockIdx.x * blockDim.x + threadIdx.x;
    if (idx >= M * HID) return;
    int seq = idx / HID, j = idx % HID;
    const float* g = GH + seq * GN;
    float xr = g[j],        xz = g[200 + j],  xn = g[400 + j];
    float hr = g[608 + j],  hz = g[808 + j],  hn = g[1008 + j];
    float r = sigmoidf_(xr + hr);
    float z = sigmoidf_(xz + hz);
    float n = tanhf(xn + r * hn);
    float hnew = (1.0f - z) * n + z * H[idx];
    H[idx] = hnew;
    A[seq * lda + hcol + j] = f2bf(hnew);
    if (t_eff < lens[seq]) Sacc[seq * 400 + dir * HID + j] += hnew;
}
// pooled word states / len -> bf16 sentence-level input (zero-padded to 416)
__global__ void build_xs(const float* __restrict__ Sacc, const int* __restrict__ lens,
                         unsigned short* __restrict__ Xs) {
    int idx = blockIdx.x * blockDim.x + threadIdx.x;
    if (idx >= NSEQ * XS_LD) return;
    int seq = idx / XS_LD, k = idx % XS_LD;
    float v = 0.0f;
    if (k < 400) {
        int len = lens[seq];
        v = (len == 0) ? 0.0f : Sacc[seq * 400 + k] / (float)len;
    }
    Xs[idx] = f2bf(v);
}
__global__ void doc_div(const float* __restrict__ Sdoc, const int* __restrict__ dl,
                        float* __restrict__ Dv) {
    int idx = blockIdx.x * blockDim.x + threadIdx.x;
    if (idx >= NDOC * 400) return;
    int b = idx / 400;
    int l = dl[b];
    Dv[idx] = (l == 0) ? 0.0f : Sdoc[idx] / (float)l;
}
// 11 dot products per doc (1 doc head + 10 sentence heads) via wave32 reduce
__global__ void heads(const float* __restrict__ Dv, const float* __restrict__ doc_w,
                      const float* __restrict__ doc_b, const float* __restrict__ sent_w,
                      const float* __restrict__ sent_b, float* __restrict__ out,
                      float* __restrict__ SP) {
    int b = blockIdx.x, lane = threadIdx.x;
    const float* d = Dv + b * 400;
    for (int o = 0; o < 11; ++o) {
        const float* w = (o == 0) ? doc_w : (sent_w + (o - 1) * 400);
        float s = 0.0f;
        for (int k = lane; k < 400; k += 32) s += d[k] * w[k];
        for (int off = 16; off; off >>= 1) s += __shfl_down(s, off, 32);
        if (lane == 0) {
            if (o == 0) out[b] = sigmoidf_(s + doc_b[0]);
            else        SP[b * NSENT + (o - 1)] = sigmoidf_(s + sent_b[o - 1]);
        }
    }
}
// ragged gather of per-sentence outputs after the 256 doc outputs
__global__ void scatter_out(const float* __restrict__ SP, const int* __restrict__ dl,
                            float* __restrict__ out) {
    __shared__ int pre[NDOC];
    if (threadIdx.x == 0) {
        int s = 0;
        for (int i = 0; i < NDOC; ++i) { pre[i] = s; s += dl[i]; }
    }
    __syncthreads();
    int b = threadIdx.x;
    int n = dl[b], p = pre[b];
    for (int c = 0; c < n; ++c) out[NDOC + p + c] = SP[b * NSENT + c];
}

// ---------------------------------------------------------------------------
// Host launch
// ---------------------------------------------------------------------------
static inline char* bump(char*& p, size_t bytes) {
    char* r = p;
    p += (bytes + 255) & ~size_t(255);
    return r;
}

extern "C" void kernel_launch(void* const* d_in, const int* in_sizes, int n_in,
                              void* d_out, int out_size, void* d_ws, size_t ws_size,
                              hipStream_t stream) {
    (void)in_sizes; (void)n_in; (void)out_size; (void)ws_size;
    const int*   x       = (const int*)d_in[0];
    const int*   dl      = (const int*)d_in[2];
    const float* emb     = (const float*)d_in[3];
    const float* wf_ih   = (const float*)d_in[4];
    const float* wf_hh   = (const float*)d_in[5];
    const float* wf_bi   = (const float*)d_in[6];
    const float* wf_bh   = (const float*)d_in[7];
    const float* wb_ih   = (const float*)d_in[8];
    const float* wb_hh   = (const float*)d_in[9];
    const float* wb_bi   = (const float*)d_in[10];
    const float* wb_bh   = (const float*)d_in[11];
    const float* sf_ih   = (const float*)d_in[12];
    const float* sf_hh   = (const float*)d_in[13];
    const float* sf_bi   = (const float*)d_in[14];
    const float* sf_bh   = (const float*)d_in[15];
    const float* sb_ih   = (const float*)d_in[16];
    const float* sb_hh   = (const float*)d_in[17];
    const float* sb_bi   = (const float*)d_in[18];
    const float* sb_bh   = (const float*)d_in[19];
    const float* doc_w   = (const float*)d_in[20];
    const float* doc_b   = (const float*)d_in[21];
    const float* sent_w  = (const float*)d_in[22];
    const float* sent_b  = (const float*)d_in[23];
    float* out = (float*)d_out;

    // workspace layout (~30 MB)
    char* p = (char*)d_ws;
    int*            lens  = (int*)           bump(p, NSEQ * sizeof(int));
    unsigned short* WcWF  = (unsigned short*)bump(p, (size_t)LDA_W * GN * 2);
    unsigned short* WcWB  = (unsigned short*)bump(p, (size_t)LDA_W * GN * 2);
    unsigned short* WcSF  = (unsigned short*)bump(p, (size_t)LDA_S * GN * 2);
    unsigned short* WcSB  = (unsigned short*)bump(p, (size_t)LDA_S * GN * 2);
    float*          BWF   = (float*)         bump(p, GN * 4);
    float*          BWB   = (float*)         bump(p, GN * 4);
    float*          BSF   = (float*)         bump(p, GN * 4);
    float*          BSB   = (float*)         bump(p, GN * 4);
    unsigned short* Abuf  = (unsigned short*)bump(p, (size_t)NSEQ * LDA_W * 2);
    float*          GH    = (float*)         bump(p, (size_t)NSEQ * GN * 4);
    float*          H     = (float*)         bump(p, (size_t)NSEQ * HID * 4);
    float*          Sacc  = (float*)         bump(p, (size_t)NSEQ * 400 * 4);
    unsigned short* Xs    = (unsigned short*)bump(p, (size_t)NSEQ * XS_LD * 2);
    float*          Sdoc  = (float*)         bump(p, (size_t)NDOC * 400 * 4);
    float*          Dv    = (float*)         bump(p, (size_t)NDOC * 400 * 4);
    float*          SP    = (float*)         bump(p, (size_t)NDOC * NSENT * 4);

    const int TB = 256;
    #define NB(n) (((n) + TB - 1) / TB)

    compute_lens<<<NB(NSEQ), TB, 0, stream>>>(x, lens);
    fill_wc_word<<<NB(LDA_W * GN), TB, 0, stream>>>(wf_ih, wf_hh, WcWF);
    fill_wc_word<<<NB(LDA_W * GN), TB, 0, stream>>>(wb_ih, wb_hh, WcWB);
    fill_wc_sent<<<NB(LDA_S * GN), TB, 0, stream>>>(sf_ih, sf_hh, WcSF);
    fill_wc_sent<<<NB(LDA_S * GN), TB, 0, stream>>>(sb_ih, sb_hh, WcSB);
    fill_bias<<<NB(GN), TB, 0, stream>>>(wf_bi, wf_bh, BWF);
    fill_bias<<<NB(GN), TB, 0, stream>>>(wb_bi, wb_bh, BWB);
    fill_bias<<<NB(GN), TB, 0, stream>>>(sf_bi, sf_bh, BSF);
    fill_bias<<<NB(GN), TB, 0, stream>>>(sb_bi, sb_bh, BSB);
    zero_f32<<<NB(NSEQ * 400), TB, 0, stream>>>(Sacc, NSEQ * 400);
    zero_f32<<<NB(NDOC * 400), TB, 0, stream>>>(Sdoc, NDOC * 400);

    // ---- word-level BiGRU ----
    for (int dir = 0; dir < 2; ++dir) {
        zero_f32<<<NB(NSEQ * HID), TB, 0, stream>>>(H, NSEQ * HID);
        zero_u16<<<NB(NSEQ * LDA_W), TB, 0, stream>>>(Abuf, NSEQ * LDA_W);
        const unsigned short* Wc = dir ? WcWB : WcWF;
        const float*          Bb = dir ? BWB  : BWF;
        for (int t = 0; t < WORD_T; ++t) {
            int t_eff = dir ? (WORD_T - 1 - t) : t;
            gather_word<<<NB(NSEQ * EMB_D), TB, 0, stream>>>(x, emb, Abuf, t_eff);
            gemm_bf16_wmma<<<dim3(NSEQ / 128, GN / 64), TB, 0, stream>>>(
                Abuf, LDA_W, NSEQ, Wc, Bb, GH, LDA_W / 32);
            gru_gate<<<NB(NSEQ * HID), TB, 0, stream>>>(
                GH, H, Abuf, LDA_W, 320, Sacc, lens, t_eff, dir, NSEQ);
        }
    }
    build_xs<<<NB(NSEQ * XS_LD), TB, 0, stream>>>(Sacc, lens, Xs);

    // ---- sentence-level BiGRU ----
    for (int dir = 0; dir < 2; ++dir) {
        zero_f32<<<NB(NDOC * HID), TB, 0, stream>>>(H, NDOC * HID);
        zero_u16<<<NB(NDOC * LDA_S), TB, 0, stream>>>(Abuf, NDOC * LDA_S);
        const unsigned short* Wc = dir ? WcSB : WcSF;
        const float*          Bb = dir ? BSB  : BSF;
        for (int t = 0; t < NSENT; ++t) {
            int t_eff = dir ? (NSENT - 1 - t) : t;
            gather_sent<<<NB(NDOC * XS_LD), TB, 0, stream>>>(Xs, Abuf, t_eff);
            gemm_bf16_wmma<<<dim3(NDOC / 128, GN / 64), TB, 0, stream>>>(
                Abuf, LDA_S, NDOC, Wc, Bb, GH, LDA_S / 32);
            gru_gate<<<NB(NDOC * HID), TB, 0, stream>>>(
                GH, H, Abuf, LDA_S, XS_LD, Sdoc, dl, t_eff, dir, NDOC);
        }
    }

    doc_div<<<NB(NDOC * 400), TB, 0, stream>>>(Sdoc, dl, Dv);
    heads<<<NDOC, 32, 0, stream>>>(Dv, doc_w, doc_b, sent_w, sent_b, out, SP);
    scatter_out<<<1, NDOC, 0, stream>>>(SP, dl, out);
}
